// MultiplexRoutingHead_26018911879231
// MI455X (gfx1250) — matrix operand order repaired and checked by reading, not compiled, not run on hardware
//
#include <hip/hip_runtime.h>
#include <hip/hip_bf16.h>
#include <math.h>

// Problem constants
#define P_  1024
#define D_  512
#define H_  256
#define HG_ 128
#define E_  8
#define T_  5
#define Q_  2048

typedef __attribute__((ext_vector_type(16))) __bf16 v16bf;
typedef __attribute__((ext_vector_type(8)))  float  v8f;

__device__ __forceinline__ v8f wmma_bf16(v16bf a, v16bf b, v8f c) {
    // (neg_a, A, neg_b, B, c_mod, C, reuse_a, reuse_b)
    return __builtin_amdgcn_wmma_f32_16x16x32_bf16(false, a, false, b, (short)0, c, false, false);
}

// K index for element t (0..15) of a 16-bit A fragment, given lane-half hi.
// Per ISA 16-bit A layout: lanes 0-15 hold K{0..7,16..23}, lanes 16-31 hold K{8..15,24..31}.
__device__ __forceinline__ int akof(int t, int hi) {
    return ((t & 8) << 1) + hi * 8 + (t & 7);
}

// ---------------- Gate: [1,2565] -> PReLU -> LN -> [1,8] -> top2 softmax ----
__global__ __launch_bounds__(128) void gate_kernel(
    const float* __restrict__ prot, const float* __restrict__ vpr,
    const float* __restrict__ dm,   const float* __restrict__ tv,
    const float* __restrict__ Wg1,  const float* __restrict__ bg1,
    const float* __restrict__ ag1,  const float* __restrict__ lnw,
    const float* __restrict__ lnb,  const float* __restrict__ Wg2,
    const float* __restrict__ bg2,  float* __restrict__ gate_ws,
    float* __restrict__ out)
{
    __shared__ float red[128];
    __shared__ float hn[128];
    __shared__ float lg[8];
    const int t = threadIdx.x;
    const float* wrow = Wg1 + (size_t)t * (2 * P_ + D_ + T_);
    float acc = bg1[t];
    for (int i = 0; i < P_;  ++i) acc += wrow[i]            * prot[i];
    for (int i = 0; i < P_;  ++i) acc += wrow[P_ + i]       * vpr[i];
    for (int i = 0; i < D_;  ++i) acc += wrow[2 * P_ + i]   * dm[i];
    for (int i = 0; i < T_;  ++i) acc += wrow[2 * P_ + D_ + i] * tv[i];
    const float ag = ag1[0];
    const float h = (acc >= 0.f) ? acc : ag * acc;
    red[t] = h; __syncthreads();
    for (int s = 64; s > 0; s >>= 1) { if (t < s) red[t] += red[t + s]; __syncthreads(); }
    const float mu = red[0] * (1.f / 128.f);
    __syncthreads();
    const float dv = h - mu;
    red[t] = dv * dv; __syncthreads();
    for (int s = 64; s > 0; s >>= 1) { if (t < s) red[t] += red[t + s]; __syncthreads(); }
    const float var = red[0] * (1.f / 128.f);
    hn[t] = dv * rsqrtf(var + 1e-5f) * lnw[t] + lnb[t];
    __syncthreads();
    if (t < E_) {
        float l = bg2[t];
        const float* w2 = Wg2 + t * HG_;
        for (int i = 0; i < HG_; ++i) l += w2[i] * hn[i];
        lg[t] = l;
    }
    __syncthreads();
    if (t == 0) {
        int i1 = 0; float v1 = lg[0];
        for (int i = 1; i < E_; ++i) if (lg[i] > v1) { v1 = lg[i]; i1 = i; }
        int i2 = -1; float v2 = 0.f;
        for (int i = 0; i < E_; ++i) { if (i == i1) continue; if (i2 < 0 || lg[i] > v2) { v2 = lg[i]; i2 = i; } }
        const float e2  = __expf(v2 - v1);
        const float inv = 1.f / (1.f + e2);
        for (int i = 0; i < E_; ++i) {
            const float g = (i == i1) ? inv : ((i == i2) ? e2 * inv : 0.f);
            gate_ws[i] = g;
            out[Q_ + i] = g;
        }
    }
}

// ------- p[e,h] = Wp[e,h,:]·xp + bp ; dconst[e,h] = Wd[e,h,512:]·[vpr,dm] + bd
__global__ __launch_bounds__(256) void pd_kernel(
    const float* __restrict__ z,  const float* __restrict__ vpr,
    const float* __restrict__ dm, const float* __restrict__ Wp,
    const float* __restrict__ bp, const float* __restrict__ Wd,
    const float* __restrict__ bd, float* __restrict__ p_ws,
    float* __restrict__ dc_ws)
{
    __shared__ float red[256];
    const int eh = blockIdx.x;          // e*H + h
    const int t  = threadIdx.x;
    const float* wp = Wp + (size_t)eh * (2 * P_ + D_);
    float acc = 0.f;
    for (int i = t; i < 2 * P_ + D_; i += 256) {
        const float x = (i < P_) ? z[i] : ((i < 2 * P_) ? vpr[i - P_] : dm[i - 2 * P_]);
        acc += wp[i] * x;
    }
    red[t] = acc; __syncthreads();
    for (int s = 128; s > 0; s >>= 1) { if (t < s) red[t] += red[t + s]; __syncthreads(); }
    if (t == 0) p_ws[eh] = red[0] + bp[eh];
    __syncthreads();
    const float* wd = Wd + (size_t)eh * (2 * D_ + P_);
    acc = 0.f;
    for (int i = D_ + t; i < 2 * D_ + P_; i += 256) {
        const float x = (i < D_ + P_) ? vpr[i - D_] : dm[i - D_ - P_];
        acc += wd[i] * x;
    }
    red[t] = acc; __syncthreads();
    for (int s = 128; s > 0; s >>= 1) { if (t < s) red[t] += red[t + s]; __syncthreads(); }
    if (t == 0) dc_ws[eh] = red[0] + bd[eh];
}

// ------- M[e,h,j] = sum_i p[e,i] * Wb[e,h,i,j]   (streams all 537 MB of Wb)
// b128 streaming: thread owns a float4 column group, strides i by 4; then a
// 4-way LDS reduction folds the i-subsets together.
__global__ __launch_bounds__(256) void mproj_kernel(
    const float* __restrict__ Wb, const float* __restrict__ p_ws,
    float* __restrict__ M_ws)
{
    __shared__ float ps[H_];
    __shared__ float red[4][H_];        // [i-subset][j]
    const int eh = blockIdx.x;          // e*H + h
    const int e  = eh >> 8;
    const int t  = threadIdx.x;
    ps[t] = p_ws[e * H_ + t];
    __syncthreads();
    const int isub = t >> 6;            // 0..3 (wave-uniform)
    const int j4   = (t & 63) * 4;      // 16B-aligned column base
    const float* wb = Wb + (size_t)eh * H_ * H_;  // [i][j]
    float4 acc = make_float4(0.f, 0.f, 0.f, 0.f);
#pragma unroll 4
    for (int i = isub; i < H_; i += 4) {
        const float4 w = *(const float4*)(wb + (size_t)i * H_ + j4);
        const float pv = ps[i];
        acc.x += pv * w.x; acc.y += pv * w.y;
        acc.z += pv * w.z; acc.w += pv * w.w;
    }
    *(float4*)&red[isub][j4] = acc;
    __syncthreads();
    M_ws[(size_t)eh * H_ + t] = red[0][t] + red[1][t] + red[2][t] + red[3][t];
}

// ------- d[q,h] = query[q,:512] · Wd[e,h,:512] + dconst[e,h]   (WMMA bf16)
__global__ __launch_bounds__(256) void dgemm_kernel(
    const float* __restrict__ query,   // [Q, D_]
    const float* __restrict__ Wd,      // [E, H, 2D+P]
    const float* __restrict__ dconst,  // [E*H]
    float* __restrict__ dbuf,          // [Q, H] (current expert)
    int e)
{
    const int wave = threadIdx.x >> 5;
    const int lane = threadIdx.x & 31;
    const int tile = blockIdx.x * 8 + wave;     // (Q/16)*(H/16) = 2048 tiles
    const int qt = tile >> 4;
    const int ht = tile & 15;
    const int q0 = qt * 16, h0 = ht * 16;
    const int m  = lane & 15;
    const int hi = lane >> 4;
    v8f c = {};
    const float* arow = query + (size_t)(q0 + m) * D_;
    const float* brow = Wd + ((size_t)e * H_ + h0 + m) * (2 * D_ + P_);
    for (int k0 = 0; k0 < D_; k0 += 32) {
        v16bf a, b;
#pragma unroll
        for (int t = 0; t < 16; ++t) a[t] = (__bf16)arow[k0 + akof(t, hi)];
#pragma unroll
        for (int t = 0; t < 16; ++t) b[t] = (__bf16)brow[k0 + akof(t, hi)];
        c = wmma_bf16(a, b, c);
    }
    const float dc = dconst[e * H_ + h0 + m];
#pragma unroll
    for (int v = 0; v < 8; ++v)
        dbuf[(size_t)(q0 + v + 8 * hi) * H_ + h0 + m] = c[v] + dc;
}

// ------- fused: t = d·M^T (+bb, PReLU) -> h2 = h1·Wm1^T (+bm1, PReLU) -> score
__global__ __launch_bounds__(64) void expert_tail_kernel(
    const float* __restrict__ dbuf,   // [Q, H] current expert
    const float* __restrict__ Mw,     // [E, H, H]
    const float* __restrict__ bb,     // [E, H]
    const float* __restrict__ a1,     // [E]
    const float* __restrict__ Wm1,    // [E, 128, 256]
    const float* __restrict__ bm1,    // [E, 128]
    const float* __restrict__ a2,     // [E]
    const float* __restrict__ Wm2,    // [E, 128]
    const float* __restrict__ bm2,    // [E]
    float* __restrict__ out_expert,   // [Q, E]
    int e)
{
    __shared__ float h1s[2][16][261];  // odd stride: conflict-free rows
    __shared__ float h2s[2][16][137];
    const int wave = threadIdx.x >> 5;
    const int lane = threadIdx.x & 31;
    const int m  = lane & 15;
    const int hi = lane >> 4;
    const int q0 = (blockIdx.x * 2 + wave) * 16;

    // preload A fragments of d tile (16 x 256), 8 k-steps
    v16bf afr[8];
    {
        const float* drow = dbuf + (size_t)(q0 + m) * H_;
#pragma unroll
        for (int ks = 0; ks < 8; ++ks)
#pragma unroll
            for (int t = 0; t < 16; ++t)
                afr[ks][t] = (__bf16)drow[ks * 32 + akof(t, hi)];
    }
    const float aL1 = a1[e];
    for (int ht = 0; ht < 16; ++ht) {
        v8f c = {};
        const float* mrow = Mw + ((size_t)e * H_ + ht * 16 + m) * H_;
#pragma unroll
        for (int ks = 0; ks < 8; ++ks) {
            v16bf b;
#pragma unroll
            for (int t = 0; t < 16; ++t) b[t] = (__bf16)mrow[ks * 32 + akof(t, hi)];
            c = wmma_bf16(afr[ks], b, c);
        }
        const float bbv = bb[e * H_ + ht * 16 + m];
#pragma unroll
        for (int v = 0; v < 8; ++v) {
            float x = c[v] + bbv;
            x = (x >= 0.f) ? x : aL1 * x;
            h1s[wave][v + 8 * hi][ht * 16 + m] = x;
        }
    }
    // rebuild A fragments from h1 in LDS (same-wave LDS ordering is in-order)
    v16bf hfr[8];
#pragma unroll
    for (int ks = 0; ks < 8; ++ks)
#pragma unroll
        for (int t = 0; t < 16; ++t)
            hfr[ks][t] = (__bf16)h1s[wave][m][ks * 32 + akof(t, hi)];

    const float aL2 = a2[e];
    for (int ot = 0; ot < 8; ++ot) {
        v8f c = {};
        const float* wrow = Wm1 + ((size_t)e * 128 + ot * 16 + m) * H_;
#pragma unroll
        for (int ks = 0; ks < 8; ++ks) {
            v16bf b;
#pragma unroll
            for (int t = 0; t < 16; ++t) b[t] = (__bf16)wrow[ks * 32 + akof(t, hi)];
            c = wmma_bf16(hfr[ks], b, c);
        }
        const float bmv = bm1[e * 128 + ot * 16 + m];
#pragma unroll
        for (int v = 0; v < 8; ++v) {
            float x = c[v] + bmv;
            x = (x >= 0.f) ? x : aL2 * x;
            h2s[wave][v + 8 * hi][ot * 16 + m] = x;
        }
    }
    // score[q] = h2[q,:] · Wm2[e] + bm2[e]
    if (lane < 16) {
        float acc = bm2[e];
        const float* w2 = Wm2 + (size_t)e * 128;
        for (int j = 0; j < 128; ++j) acc += h2s[wave][lane][j] * w2[j];
        out_expert[(size_t)(q0 + lane) * E_ + e] = acc;
    }
}

// ------- final_scores[q] = sum_e expert_tensor[q,e] * gate[e]
__global__ __launch_bounds__(256) void finalize_kernel(
    const float* __restrict__ gate, float* __restrict__ out)
{
    const int q = blockIdx.x * 256 + threadIdx.x;
    if (q >= Q_) return;
    const float* et = out + Q_ + E_ + (size_t)q * E_;
    float s = 0.f;
#pragma unroll
    for (int e = 0; e < E_; ++e) s += et[e] * gate[e];
    out[q] = s;
}

extern "C" void kernel_launch(void* const* d_in, const int* in_sizes, int n_in,
                              void* d_out, int out_size, void* d_ws, size_t ws_size,
                              hipStream_t stream) {
    (void)in_sizes; (void)n_in; (void)out_size; (void)ws_size;
    const float* z    = (const float*)d_in[0];
    const float* prot = (const float*)d_in[1];
    const float* vpr  = (const float*)d_in[2];
    const float* dm   = (const float*)d_in[3];
    const float* query= (const float*)d_in[4];
    const float* tv   = (const float*)d_in[5];
    const float* Wg1  = (const float*)d_in[6];
    const float* bg1  = (const float*)d_in[7];
    const float* ag1  = (const float*)d_in[8];
    const float* lnw  = (const float*)d_in[9];
    const float* lnb  = (const float*)d_in[10];
    const float* Wg2  = (const float*)d_in[11];
    const float* bg2  = (const float*)d_in[12];
    const float* Wp   = (const float*)d_in[13];
    const float* bp   = (const float*)d_in[14];
    const float* Wd   = (const float*)d_in[15];
    const float* bd   = (const float*)d_in[16];
    const float* Wb   = (const float*)d_in[17];
    const float* bb   = (const float*)d_in[18];
    const float* a1   = (const float*)d_in[19];
    const float* Wm1  = (const float*)d_in[20];
    const float* bm1  = (const float*)d_in[21];
    const float* a2   = (const float*)d_in[22];
    const float* Wm2  = (const float*)d_in[23];
    const float* bm2  = (const float*)d_in[24];

    float* out = (float*)d_out;
    float* ws  = (float*)d_ws;
    float* p_ws    = ws;                                   // E*H = 2048
    float* dc_ws   = ws + 2048;                            // E*H = 2048
    float* gate_ws = ws + 4096;                            // 8 (padded)
    float* M_ws    = ws + 4352;                            // E*H*H = 524288
    float* dbuf    = ws + 4352 + (size_t)E_ * H_ * H_;     // Q*H  = 524288

    gate_kernel<<<1, 128, 0, stream>>>(prot, vpr, dm, tv, Wg1, bg1, ag1, lnw, lnb,
                                       Wg2, bg2, gate_ws, out);
    pd_kernel<<<E_ * H_, 256, 0, stream>>>(z, vpr, dm, Wp, bp, Wd, bd, p_ws, dc_ws);
    mproj_kernel<<<E_ * H_, 256, 0, stream>>>(Wb, p_ws, M_ws);
    for (int e = 0; e < E_; ++e) {
        dgemm_kernel<<<256, 256, 0, stream>>>(query, Wd, dc_ws, dbuf, e);
        expert_tail_kernel<<<64, 64, 0, stream>>>(dbuf, M_ws, bb, a1, Wm1, bm1,
                                                  a2, Wm2, bm2, out + Q_ + E_, e);
    }
    finalize_kernel<<<E_, 256, 0, stream>>>(gate_ws, out);
}
